// GaugeTransformerBlock_2327872275162
// MI455X (gfx1250) — matrix operand order repaired and checked by reading, not compiled, not run on hardware
//
#include <hip/hip_runtime.h>
#include <hip/hip_bf16.h>
#include <math.h>

// ---------------- problem constants ----------------
#define BATCH 4
#define SEQ   1024
#define KD    512
#define NHEAD 8
#define DHEAD 64
#define HIDN  2048
#define TOK   (BATCH*SEQ)                 // 4096 tokens
#define BNK   ((size_t)TOK*KD)            // 2,097,152 floats per activation

#define EPS_F   1e-8f
#define ALPHA_F 1e-3f
#define LAM_F   1.0f

// ---------------- GEMM tiling ----------------
#define BM  128
#define BNL 128
#define BK  32

typedef __attribute__((ext_vector_type(16))) __bf16 v16bf;
typedef __attribute__((ext_vector_type(8)))  float  v8f;

union Frag { v16bf v; float4 f4[2]; };

enum { EP_NONE=0, EP_BIAS=1, EP_BIAS_GELU=2, EP_SCORE=3, EP_PHI=4 };

// Types matching the async-to-LDS builtin signature:
//   void __builtin_amdgcn_global_load_async_to_lds_b128(v4i AS1*, v4i AS3*, imm, imm)
typedef int v4i_t __attribute__((vector_size(16)));
typedef v4i_t __attribute__((address_space(1))) *g_v4i_p;   // global (AS1) int4*
typedef v4i_t __attribute__((address_space(3))) *l_v4i_p;   // LDS    (AS3) int4*

__device__ __forceinline__ unsigned short f2bf(float f) {
    unsigned int u = __float_as_uint(f);
    unsigned int r = (u + 0x7FFFu + ((u >> 16) & 1u)) >> 16;   // RNE
    return (unsigned short)r;
}

// 16-byte global -> LDS copy. Prefer the CDNA5 async-DMA path (ASYNCcnt,
// no VGPR round trip); fall back to a sync copy if builtin is absent.
__device__ __forceinline__ void cp16_g2l(const float* g, float* l) {
#if __has_builtin(__builtin_amdgcn_global_load_async_to_lds_b128)
    __builtin_amdgcn_global_load_async_to_lds_b128(
        (g_v4i_p)(void*)g, (l_v4i_p)(void*)l, 0, 0);
#else
    *(float4*)l = *(const float4*)g;
#endif
}

__device__ __forceinline__ void wait_async_all() {
#if __has_builtin(__builtin_amdgcn_global_load_async_to_lds_b128)
#if __has_builtin(__builtin_amdgcn_s_wait_asynccnt)
    __builtin_amdgcn_s_wait_asynccnt(0);
#else
    asm volatile("s_wait_asynccnt 0" ::: "memory");
#endif
#endif
}

// ============================================================
// Generic bf16-operand / f32-accumulate WMMA GEMM, specialized
// at compile time on (TRANSB, SQB, EP).
//   C[i,j] = epilogue( sum_c A[i,c] * Bop[c,j] ),  Bop = transB? B^T : B
// Pipeline per K-slab:
//   async DMA (global->LDS fp32, double buffered)  ->  fp32->bf16
//   convert/relayout (absorbs B transpose)         ->  8x v_wmma
// ============================================================
template<int TRANSB, int SQB, int EP>
__global__ __launch_bounds__(256)
void wmma_gemm_kernel(const float* __restrict__ Ag, const float* __restrict__ Bg,
                      float* __restrict__ Cg,
                      int M, int Nd, int Kd, int lda, int ldb, int ldc,
                      long long sAb, long long sAh,
                      long long sBb, long long sBh,
                      long long sCb, long long sCh, int innerH,
                      const float* __restrict__ aux0,
                      const float* __restrict__ aux1,
                      float scale, int accum, int phig)
{
    __shared__ __align__(16) float Af32[2][BM * BK];          // raw fp32 staging (async dest)
    __shared__ __align__(16) float Bf32[2][BK * BNL];         // raw fp32 staging (memory layout)
    __shared__ __align__(16) unsigned short As[BM][BK];       // bf16, fragment-ready
    __shared__ __align__(16) unsigned short Bs[BNL][BK];      // bf16 [n][k], fragment-ready

    const int z  = blockIdx.z;
    const int bo = z / innerH, ho = z - bo * innerH;
    const float* Ap = Ag + bo * sAb + ho * sAh;
    const float* Bp = Bg + bo * sBb + ho * sBh;
    float*       Cp = Cg + bo * sCb + ho * sCh;

    const int tid  = threadIdx.x;
    const int lane = tid & 31;
    const int wave = tid >> 5;            // 8 waves (wave32)
    const int lh   = lane >> 4;           // half-wave select
    const int ll   = lane & 15;
    const int wrow = (wave & 1) << 6;     // 0 / 64
    const int wcol = (wave >> 1) << 5;    // 0 / 32 / 64 / 96
    const int rowBase = blockIdx.y * BM;
    const int colBase = blockIdx.x * BNL;

    // ---- stage tile kt into staging buffer `buf` (async) ----
    auto stage = [&](int kt, int buf) {
#pragma unroll
        for (int t = 0; t < 4; ++t) {                 // A: 1024 x 16B units
            int idx = tid + t * 256;
            int r   = idx >> 3;
            int c4  = (idx & 7) << 2;
            int gr  = rowBase + r;
            if (gr < M)
                cp16_g2l(Ap + (long long)gr * lda + kt + c4, &Af32[buf][r * BK + c4]);
        }
#pragma unroll
        for (int t = 0; t < 4; ++t) {                 // B: 1024 x 16B units
            int idx = tid + t * 256;
            if (TRANSB) {                             // memory [n][k]
                int n  = idx >> 3;
                int kq = (idx & 7) << 2;
                int gn = colBase + n;
                if (gn < Nd)
                    cp16_g2l(Bp + (long long)gn * ldb + kt + kq, &Bf32[buf][n * BK + kq]);
            } else {                                  // memory [k][n]
                int kr = idx >> 5;
                int n4 = (idx & 31) << 2;
                int gn = colBase + n4;
                if (gn < Nd)
                    cp16_g2l(Bp + (long long)(kt + kr) * ldb + gn, &Bf32[buf][kr * BNL + n4]);
            }
        }
    };

    // ---- fp32 -> bf16 convert + relayout into fragment-ready tiles ----
    auto convert = [&](int buf) {
#pragma unroll
        for (int t = 0; t < 4; ++t) {
            int idx = tid + t * 256;
            int r   = idx >> 3;
            int c4  = (idx & 7) << 2;
            int gr  = rowBase + r;
            float4 v4 = make_float4(0.f, 0.f, 0.f, 0.f);
            if (gr < M) v4 = *(const float4*)&Af32[buf][r * BK + c4];
            uint2 pk;
            pk.x = (unsigned)f2bf(v4.x) | ((unsigned)f2bf(v4.y) << 16);
            pk.y = (unsigned)f2bf(v4.z) | ((unsigned)f2bf(v4.w) << 16);
            *(uint2*)&As[r][c4] = pk;
        }
#pragma unroll
        for (int t = 0; t < 4; ++t) {
            int idx = tid + t * 256;
            int n   = idx >> 3;
            int kq  = (idx & 7) << 2;
            int gn  = colBase + n;
            float4 v4 = make_float4(0.f, 0.f, 0.f, 0.f);
            if (gn < Nd) {
                if (TRANSB) {
                    v4 = *(const float4*)&Bf32[buf][n * BK + kq];
                } else {                              // transpose out of staging
                    v4.x = Bf32[buf][(kq + 0) * BNL + n];
                    v4.y = Bf32[buf][(kq + 1) * BNL + n];
                    v4.z = Bf32[buf][(kq + 2) * BNL + n];
                    v4.w = Bf32[buf][(kq + 3) * BNL + n];
                }
            }
            if (SQB) { v4.x *= v4.x; v4.y *= v4.y; v4.z *= v4.z; v4.w *= v4.w; }
            uint2 pk;
            pk.x = (unsigned)f2bf(v4.x) | ((unsigned)f2bf(v4.y) << 16);
            pk.y = (unsigned)f2bf(v4.z) | ((unsigned)f2bf(v4.w) << 16);
            *(uint2*)&Bs[n][kq] = pk;
        }
    };

    v8f acc[4][2];
#pragma unroll
    for (int i = 0; i < 4; ++i)
#pragma unroll
        for (int j = 0; j < 2; ++j)
#pragma unroll
            for (int r = 0; r < 8; ++r) acc[i][j][r] = 0.f;

    stage(0, 0);                                       // prologue DMA
    int buf = 0;
    for (int kt = 0; kt < Kd; kt += BK, buf ^= 1) {
        wait_async_all();                              // tile kt landed in LDS
        __syncthreads();                               // + all waves done with prior As/Bs
        convert(buf);
        __syncthreads();                               // As/Bs ready
        if (kt + BK < Kd) stage(kt + BK, buf ^ 1);     // DMA next slab under compute

        // ---- fragments per ISA 7.12.2 layouts (two ds_load_b128 each) ----
        Frag af[4], bfm[2];
#pragma unroll
        for (int i = 0; i < 4; ++i) {
            int r = wrow + i * 16 + ll;
            af[i].f4[0] = *(const float4*)&As[r][lh * 8];
            af[i].f4[1] = *(const float4*)&As[r][16 + lh * 8];
        }
#pragma unroll
        for (int j = 0; j < 2; ++j) {
            int n = wcol + j * 16 + ll;
            bfm[j].f4[0] = *(const float4*)&Bs[n][lh * 8];
            bfm[j].f4[1] = *(const float4*)&Bs[n][16 + lh * 8];
        }
#pragma unroll
        for (int i = 0; i < 4; ++i)
#pragma unroll
            for (int j = 0; j < 2; ++j)
                acc[i][j] = __builtin_amdgcn_wmma_f32_16x16x32_bf16(
                    false, af[i].v, false, bfm[j].v, (short)0, acc[i][j], false, false);
    }

    // ---- fused epilogue + store ----
#pragma unroll
    for (int i = 0; i < 4; ++i) {
#pragma unroll
        for (int j = 0; j < 2; ++j) {
#pragma unroll
            for (int r = 0; r < 8; ++r) {
                int grow = rowBase + wrow + i * 16 + lh * 8 + r;
                int gcol = colBase + wcol + j * 16 + ll;
                if (grow < M && gcol < Nd) {
                    float vv = acc[i][j][r];
                    long long cidx = (long long)grow * ldc + gcol;
                    float outv;
                    if (EP == EP_PHI) {               // gauge transport term
                        outv = scale * aux0[(long long)grow * 3 + phig] * vv;
                        if (accum) outv += Cp[cidx];
                    } else if (EP == EP_BIAS) {
                        outv = vv + aux0[gcol];
                    } else if (EP == EP_BIAS_GELU) {
                        float x = vv + aux0[gcol];
                        outv = 0.5f * x * (1.f + tanhf(0.7978845608028654f *
                                       (x + 0.044715f * x * x * x)));
                    } else if (EP == EP_SCORE) {      // -KL/kappa + causal mask
                        float sc = -0.5f * (vv + aux0[(long long)z * M + grow]
                                               + aux1[(long long)z * Nd + gcol]);
                        outv = (gcol > grow) ? -1e9f : sc;
                    } else {
                        outv = vv;
                        if (accum) outv += Cp[cidx];
                    }
                    Cp[cidx] = outv;
                }
            }
        }
    }
}

// ============================================================
// LayerNorm over K=512 per token
// ============================================================
__global__ __launch_bounds__(256)
void ln_kernel(const float* __restrict__ x, const float* __restrict__ g,
               const float* __restrict__ be, float* __restrict__ y)
{
    __shared__ float red[256];
    const int t = blockIdx.x;
    const float* xp = x + (long long)t * KD;
    float s = 0.f;
    for (int i = threadIdx.x; i < KD; i += 256) s += xp[i];
    red[threadIdx.x] = s; __syncthreads();
    for (int o = 128; o > 0; o >>= 1) {
        if (threadIdx.x < o) red[threadIdx.x] += red[threadIdx.x + o];
        __syncthreads();
    }
    float mean = red[0] * (1.f / KD);
    __syncthreads();
    float v = 0.f;
    for (int i = threadIdx.x; i < KD; i += 256) { float d = xp[i] - mean; v += d * d; }
    red[threadIdx.x] = v; __syncthreads();
    for (int o = 128; o > 0; o >>= 1) {
        if (threadIdx.x < o) red[threadIdx.x] += red[threadIdx.x + o];
        __syncthreads();
    }
    float rstd = rsqrtf(red[0] * (1.f / KD) + 1e-5f);
    for (int i = threadIdx.x; i < KD; i += 256)
        y[(long long)t * KD + i] = (xp[i] - mean) * rstd * g[i] + be[i];
}

// out = resid + x + a1 + 0.5*a2   (resid may be null)
__global__ void transport_finish_kernel(const float* __restrict__ x,
                                        const float* __restrict__ a1,
                                        const float* __restrict__ a2,
                                        const float* __restrict__ resid,
                                        float* __restrict__ out, int n)
{
    int i = blockIdx.x * 256 + threadIdx.x;
    if (i < n) {
        float r = resid ? resid[i] : 0.f;
        out[i] = r + x[i] + a1[i] + 0.5f * a2[i];
    }
}

// Build P=[inv,-2 q inv, inv], Q=[sk,k,k^2] and row/col log-det terms
__global__ __launch_bounds__(64)
void prep_attn_kernel(const float* __restrict__ q, const float* __restrict__ k,
                      const float* __restrict__ sqb, const float* __restrict__ skb,
                      float* __restrict__ P, float* __restrict__ Qm,
                      float* __restrict__ rowv, float* __restrict__ colv)
{
    __shared__ float r1[64], r2[64];
    const int zn = blockIdx.x;                  // (b*H+h)*SEQ + n
    const int d  = threadIdx.x;
    const int zz = zn / SEQ, n = zn - zz * SEQ;
    const int b  = zz / NHEAD, h = zz - b * NHEAD;
    const long long base = ((long long)(b * SEQ + n)) * KD + h * DHEAD + d;
    float sq  = sqb[base] + EPS_F;
    float inv = 1.f / sq;
    float qv  = q[base];
    float kv  = k[base];
    float sk  = skb[base] + EPS_F;
    const long long pb = (long long)zn * (3 * DHEAD) + d;
    P[pb]             = inv;
    P[pb + DHEAD]     = -2.f * qv * inv;
    P[pb + 2 * DHEAD] = inv;
    Qm[pb]             = sk;
    Qm[pb + DHEAD]     = kv;
    Qm[pb + 2 * DHEAD] = kv * kv;
    r1[d] = qv * qv * inv + logf(sq);
    r2[d] = logf(sk);
    __syncthreads();
    for (int o = 32; o > 0; o >>= 1) {
        if (d < o) { r1[d] += r1[d + o]; r2[d] += r2[d + o]; }
        __syncthreads();
    }
    if (d == 0) { rowv[zn] = r1[0] - (float)DHEAD; colv[zn] = -r2[0]; }
}

// row softmax over N=1024 (in place; causal mask already applied as -1e9)
__global__ __launch_bounds__(256)
void softmax_kernel(float* __restrict__ S)
{
    __shared__ float red[256];
    float* p = S + (long long)blockIdx.x * SEQ;
    float m = -1e30f;
    for (int i = threadIdx.x; i < SEQ; i += 256) m = fmaxf(m, p[i]);
    red[threadIdx.x] = m; __syncthreads();
    for (int o = 128; o > 0; o >>= 1) {
        if (threadIdx.x < o) red[threadIdx.x] = fmaxf(red[threadIdx.x], red[threadIdx.x + o]);
        __syncthreads();
    }
    float mx = red[0]; __syncthreads();
    float s = 0.f;
    for (int i = threadIdx.x; i < SEQ; i += 256) {
        float e = __expf(p[i] - mx); p[i] = e; s += e;
    }
    red[threadIdx.x] = s; __syncthreads();
    for (int o = 128; o > 0; o >>= 1) {
        if (threadIdx.x < o) red[threadIdx.x] += red[threadIdx.x + o];
        __syncthreads();
    }
    float invs = 1.f / red[0];
    for (int i = threadIdx.x; i < SEQ; i += 256) p[i] *= invs;
}

// beta_m = mean over heads
__global__ void betam_kernel(const float* __restrict__ S, float* __restrict__ bm)
{
    long long i = (long long)blockIdx.x * 256 + threadIdx.x;
    if (i < (long long)BATCH * SEQ * SEQ) {
        int b = (int)(i / ((long long)SEQ * SEQ));
        long long r = i - (long long)b * SEQ * SEQ;
        float s = 0.f;
#pragma unroll
        for (int h = 0; h < NHEAD; ++h)
            s += S[((long long)(b * NHEAD + h)) * SEQ * SEQ + r];
        bm[i] = s * (1.f / NHEAD);
    }
}

// h += lr*(drive - alpha*(h - prior) - lam*(h - coup))
__global__ void ffn_update_kernel(float* __restrict__ h, const float* __restrict__ drive,
                                  const float* __restrict__ coup,
                                  const float* __restrict__ prior,
                                  const float* __restrict__ lrp, int n)
{
    int i = blockIdx.x * 256 + threadIdx.x;
    if (i < n) {
        float lr = lrp[0];
        float hv = h[i];
        float grad = ALPHA_F * (hv - prior[i]) + LAM_F * (hv - coup[i]);
        h[i] = hv + lr * (drive[i] - grad);
    }
}

// mu_out = mu_res + h - mu_n2
__global__ void final_kernel(const float* __restrict__ mu_res, const float* __restrict__ h,
                             const float* __restrict__ mu_n2, float* __restrict__ out, int n)
{
    int i = blockIdx.x * 256 + threadIdx.x;
    if (i < n) out[i] = mu_res[i] + h[i] - mu_n2[i];
}

// ---------------- host-side launch helper (compile-time specialized) ----------------
template<int TB, int SQ, int EP>
static void launch_gemm(hipStream_t stream, const float* A, const float* Bm, float* C,
                        int M, int Nd, int Kd, int lda, int ldb, int ldc,
                        long long sAb, long long sAh, long long sBb, long long sBh,
                        long long sCb, long long sCh, int innerH, int batches,
                        const float* a0, const float* a1,
                        float scale, int accum, int phig)
{
    dim3 grid((Nd + BNL - 1) / BNL, (M + BM - 1) / BM, batches);
    wmma_gemm_kernel<TB, SQ, EP><<<grid, 256, 0, stream>>>(A, Bm, C, M, Nd, Kd,
        lda, ldb, ldc, sAb, sAh, sBb, sBh, sCb, sCh, innerH,
        a0, a1, scale, accum, phig);
}

extern "C" void kernel_launch(void* const* d_in, const int* in_sizes, int n_in,
                              void* d_out, int out_size, void* d_ws, size_t ws_size,
                              hipStream_t stream)
{
    (void)in_sizes; (void)n_in; (void)out_size; (void)ws_size;
    const float* mu_q    = (const float*)d_in[0];
    const float* sigma_q = (const float*)d_in[1];
    const float* phi     = (const float*)d_in[2];
    const float* G       = (const float*)d_in[3];   // (3,K,K)
    /* d_in[4] = mask: causal tril, applied analytically */
    const float* mu_prior= (const float*)d_in[5];
    const float* Wq = (const float*)d_in[6];
    const float* Wk = (const float*)d_in[7];
    const float* Wv = (const float*)d_in[8];
    const float* Wo = (const float*)d_in[9];
    const float* g1 = (const float*)d_in[10];
    const float* be1= (const float*)d_in[11];
    const float* g2 = (const float*)d_in[12];
    const float* be2= (const float*)d_in[13];
    const float* W1f= (const float*)d_in[14];
    const float* bh1= (const float*)d_in[15];
    const float* W2f= (const float*)d_in[16];
    const float* bh2= (const float*)d_in[17];
    const float* lr = (const float*)d_in[18];
    float* out = (float*)d_out;

    // ---------------- workspace layout ----------------
    float* ws = (float*)d_ws;
    size_t off = 0;
    float* S    = ws + off; off += (size_t)BATCH * NHEAD * SEQ * SEQ;   // scores/beta
    float* P    = ws + off; off += (size_t)BATCH * NHEAD * SEQ * 3 * DHEAD;
    float* Qm   = ws + off; off += (size_t)BATCH * NHEAD * SEQ * 3 * DHEAD;
    float* rowv = ws + off; off += (size_t)BATCH * NHEAD * SEQ;
    float* colv = ws + off; off += (size_t)BATCH * NHEAD * SEQ;
    float* betam= ws + off; off += (size_t)BATCH * SEQ * SEQ;
    float* hid  = ws + off; off += (size_t)TOK * HIDN;
    float* W0 = ws + off; off += BNK;   // mu_n -> wo_out
    float* W1 = ws + off; off += BNK;   // a1
    float* W2 = ws + off; off += BNK;   // a2
    float* W3 = ws + off; off += BNK;   // mu_g -> mu_res
    float* W4 = ws + off; off += BNK;   // q -> mu_n2
    float* W5 = ws + off; off += BNK;   // k -> h
    float* W6 = ws + off; off += BNK;   // v -> drive
    float* W7 = ws + off; off += BNK;   // sq -> coup
    float* W8 = ws + off; off += BNK;   // sk
    float* W9 = ws + off; off += BNK;   // attn_out

    const int nTok = (int)BNK;
    const int ewB  = (nTok + 255) / 256;

    // ---- 1) LN1 ----
    ln_kernel<<<TOK, 256, 0, stream>>>(mu_q, g1, be1, W0);

    // ---- 2) transport(mu_n, +phi): a1 = sum_g phi_g * (X G_g^T) ----
    for (int g = 0; g < 3; ++g)
        launch_gemm<1,0,EP_PHI>(stream, W0, G + (size_t)g * KD * KD, W1, TOK, KD, KD,
                    KD, KD, KD, 0,0,0,0,0,0, 1, 1, phi, nullptr, 1.f, (g > 0), g);
    for (int g = 0; g < 3; ++g)
        launch_gemm<1,0,EP_PHI>(stream, W1, G + (size_t)g * KD * KD, W2, TOK, KD, KD,
                    KD, KD, KD, 0,0,0,0,0,0, 1, 1, phi, nullptr, 1.f, (g > 0), g);
    transport_finish_kernel<<<ewB, 256, 0, stream>>>(W0, W1, W2, nullptr, W3, nTok);

    // ---- 3) Q,K,V and diagonal-covariance propagation ----
    launch_gemm<0,0,EP_NONE>(stream, W3, Wq, W4, TOK, KD, KD, KD, KD, KD,
                0,0,0,0,0,0,1,1, nullptr,nullptr,1.f,0,0);
    launch_gemm<0,0,EP_NONE>(stream, W3, Wk, W5, TOK, KD, KD, KD, KD, KD,
                0,0,0,0,0,0,1,1, nullptr,nullptr,1.f,0,0);
    launch_gemm<0,0,EP_NONE>(stream, W3, Wv, W6, TOK, KD, KD, KD, KD, KD,
                0,0,0,0,0,0,1,1, nullptr,nullptr,1.f,0,0);
    launch_gemm<0,1,EP_NONE>(stream, sigma_q, Wq, W7, TOK, KD, KD, KD, KD, KD,
                0,0,0,0,0,0,1,1, nullptr,nullptr,1.f,0,0);
    launch_gemm<0,1,EP_NONE>(stream, sigma_q, Wk, W8, TOK, KD, KD, KD, KD, KD,
                0,0,0,0,0,0,1,1, nullptr,nullptr,1.f,0,0);

    // ---- 4) KL-score prep and batched score GEMM (Kdim = 3*DH = 192) ----
    prep_attn_kernel<<<BATCH * NHEAD * SEQ, 64, 0, stream>>>(W4, W5, W7, W8, P, Qm, rowv, colv);
    launch_gemm<1,0,EP_SCORE>(stream, P, Qm, S, SEQ, SEQ, 3 * DHEAD,
                3 * DHEAD, 3 * DHEAD, SEQ,
                (long long)SEQ * 3 * DHEAD, 0, (long long)SEQ * 3 * DHEAD, 0,
                (long long)SEQ * SEQ, 0, 1, BATCH * NHEAD,
                rowv, colv, 1.f, 0, 0);

    // ---- 5) softmax + head-mean ----
    softmax_kernel<<<BATCH * NHEAD * SEQ, 256, 0, stream>>>(S);
    betam_kernel<<<(int)(((long long)BATCH * SEQ * SEQ + 255) / 256), 256, 0, stream>>>(S, betam);

    // ---- 6) out = beta @ v (batched over b,h; N = DH = 64) ----
    launch_gemm<0,0,EP_NONE>(stream, S, W6, W9, SEQ, DHEAD, SEQ, SEQ, KD, KD,
                (long long)NHEAD * SEQ * SEQ, (long long)SEQ * SEQ,
                (long long)SEQ * KD, DHEAD,
                (long long)SEQ * KD, DHEAD, NHEAD, BATCH * NHEAD,
                nullptr, nullptr, 1.f, 0, 0);

    // ---- 7) Wo projection, back-transport (-phi), residual ----
    launch_gemm<0,0,EP_NONE>(stream, W9, Wo, W0, TOK, KD, KD, KD, KD, KD,
                0,0,0,0,0,0,1,1, nullptr,nullptr,1.f,0,0);
    for (int g = 0; g < 3; ++g)
        launch_gemm<1,0,EP_PHI>(stream, W0, G + (size_t)g * KD * KD, W1, TOK, KD, KD,
                    KD, KD, KD, 0,0,0,0,0,0, 1, 1, phi, nullptr, -1.f, (g > 0), g);
    for (int g = 0; g < 3; ++g)
        launch_gemm<1,0,EP_PHI>(stream, W1, G + (size_t)g * KD * KD, W2, TOK, KD, KD,
                    KD, KD, KD, 0,0,0,0,0,0, 1, 1, phi, nullptr, -1.f, (g > 0), g);
    transport_finish_kernel<<<ewB, 256, 0, stream>>>(W0, W1, W2, mu_q, W3, nTok); // mu_res

    // ---- 8) VFE FFN ----
    ln_kernel<<<TOK, 256, 0, stream>>>(W3, g2, be2, W4);                          // mu_n2
    (void)hipMemcpyAsync(W5, W4, BNK * sizeof(float), hipMemcpyDeviceToDevice, stream); // h = mu_n2
    for (int it = 0; it < 2; ++it) {
        launch_gemm<0,0,EP_BIAS_GELU>(stream, W5, W1f, hid, TOK, HIDN, KD,
                    KD, HIDN, HIDN, 0,0,0,0,0,0,1,1, bh1, nullptr, 1.f, 0, 0);
        launch_gemm<0,0,EP_BIAS>(stream, hid, W2f, W6, TOK, KD, HIDN,
                    HIDN, KD, KD, 0,0,0,0,0,0,1,1, bh2, nullptr, 1.f, 0, 0);
        launch_gemm<0,0,EP_NONE>(stream, betam, W5, W7, SEQ, KD, SEQ, SEQ, KD, KD,
                    (long long)SEQ * SEQ, 0, (long long)SEQ * KD, 0,
                    (long long)SEQ * KD, 0, 1, BATCH,
                    nullptr, nullptr, 1.f, 0, 0);
        ffn_update_kernel<<<ewB, 256, 0, stream>>>(W5, W6, W7, mu_prior, lr, nTok);
    }

    // ---- 9) outputs: (mu_out, sigma_q, phi) concatenated ----
    final_kernel<<<ewB, 256, 0, stream>>>(W3, W5, W4, out, nTok);
    (void)hipMemcpyAsync(out + BNK, sigma_q, BNK * sizeof(float), hipMemcpyDeviceToDevice, stream);
    (void)hipMemcpyAsync(out + 2 * BNK, phi, (size_t)BATCH * SEQ * 3 * sizeof(float),
                   hipMemcpyDeviceToDevice, stream);
}
